// MOELayer_13804024889477
// MI455X (gfx1250) — matrix operand order repaired and checked by reading, not compiled, not run on hardware
//
#include <hip/hip_runtime.h>

// Problem dims (match reference)
#define B_DIM 4096
#define D_IN  1024
#define D_HID 512
#define D_OUT 1024
#define E_NUM 16
#define D_CTX 128

typedef __attribute__((ext_vector_type(16))) __bf16 v16bf;
typedef __attribute__((ext_vector_type(8)))  float  v8f;

union Frag16 { v16bf v; uint4 q[2]; };

__device__ __forceinline__ unsigned short f32_to_bf16(float f) {
    unsigned int u = __float_as_uint(f);
    u += 0x7FFFu + ((u >> 16) & 1u);   // round-to-nearest-even
    return (unsigned short)(u >> 16);
}

// ---------------------------------------------------------------------------
// Gate: logits = concat(u,x) @ Wg + bg ; softmax over E. One thread per row.
// ---------------------------------------------------------------------------
__global__ __launch_bounds__(256) void gate_kernel(
    const float* __restrict__ x, const float* __restrict__ u,
    const float* __restrict__ Wg, const float* __restrict__ bg,
    float* __restrict__ wts)
{
    int b = blockIdx.x * blockDim.x + threadIdx.x;
    if (b >= B_DIM) return;
    float logit[E_NUM];
#pragma unroll
    for (int e = 0; e < E_NUM; ++e) logit[e] = bg[e];
    const float* ur = u + (size_t)b * D_CTX;
    const float* xr = x + (size_t)b * D_IN;
    for (int k = 0; k < D_CTX; ++k) {
        float gv = ur[k];
        const float* wrow = Wg + (size_t)k * E_NUM;
#pragma unroll
        for (int e = 0; e < E_NUM; ++e) logit[e] = fmaf(gv, wrow[e], logit[e]);
    }
    for (int k = 0; k < D_IN; ++k) {
        float gv = xr[k];
        const float* wrow = Wg + (size_t)(D_CTX + k) * E_NUM;
#pragma unroll
        for (int e = 0; e < E_NUM; ++e) logit[e] = fmaf(gv, wrow[e], logit[e]);
    }
    float m = logit[0];
#pragma unroll
    for (int e = 1; e < E_NUM; ++e) m = fmaxf(m, logit[e]);
    float s = 0.f;
#pragma unroll
    for (int e = 0; e < E_NUM; ++e) { logit[e] = __expf(logit[e] - m); s += logit[e]; }
    float inv = 1.f / s;
#pragma unroll
    for (int e = 0; e < E_NUM; ++e) wts[(size_t)b * E_NUM + e] = logit[e] * inv;
}

// ---------------------------------------------------------------------------
// f32 -> bf16 bulk conversion (row-major preserved), 4 elements / thread
// ---------------------------------------------------------------------------
__global__ __launch_bounds__(256) void convert_kernel(
    const float* __restrict__ src, unsigned short* __restrict__ dst, int n4)
{
    int i = blockIdx.x * blockDim.x + threadIdx.x;
    if (i >= n4) return;
    float4 f = ((const float4*)src)[i];
    uint2 o;
    o.x = (unsigned int)f32_to_bf16(f.x) | ((unsigned int)f32_to_bf16(f.y) << 16);
    o.y = (unsigned int)f32_to_bf16(f.z) | ((unsigned int)f32_to_bf16(f.w) << 16);
    ((uint2*)dst)[i] = o;
}

// ---------------------------------------------------------------------------
// f32 -> bf16 transpose-convert: in [E][K][N] f32 -> out [E][N][K] bf16.
// LDS-tiled 32x32, block (32,8).
// ---------------------------------------------------------------------------
__global__ __launch_bounds__(256) void transpose_convert_kernel(
    const float* __restrict__ src, unsigned short* __restrict__ dst,
    int K, int N)
{
    __shared__ float tile[32][33];
    const int e  = blockIdx.z;
    const int k0 = blockIdx.y * 32;
    const int n0 = blockIdx.x * 32;
    const int tx = threadIdx.x, ty = threadIdx.y;
    const float* s = src + (size_t)e * K * N;
#pragma unroll
    for (int j = 0; j < 4; ++j)
        tile[ty + j * 8][tx] = s[(size_t)(k0 + ty + j * 8) * N + n0 + tx];
    __syncthreads();
    unsigned short* d = dst + (size_t)e * N * K;
#pragma unroll
    for (int j = 0; j < 4; ++j)
        d[(size_t)(n0 + ty + j * 8) * K + k0 + tx] = f32_to_bf16(tile[tx][ty + j * 8]);
}

// ---------------------------------------------------------------------------
// Async tile staging: issue 4x b128 global->LDS DMA per thread (one 128x64
// bf16 tile per 256-thread block). Per-thread offsets precomputed; stage
// base folded into the scalar operand (SALU), buffer offset one v_add.
// ---------------------------------------------------------------------------
__device__ __forceinline__ void issue4(
    const unsigned short* gbaseStage, const unsigned* pvoff,
    const unsigned* ldsChunk, unsigned bufOff)
{
#pragma unroll
    for (int i = 0; i < 4; ++i)
        asm volatile("global_load_async_to_lds_b128 %0, %1, %2"
                     :: "v"(ldsChunk[i] + bufOff), "v"(pvoff[i]), "s"(gbaseStage)
                     : "memory");
}

__device__ __forceinline__ void wait_async0() {
    asm volatile("s_wait_asynccnt 0" ::: "memory");
}
__device__ __forceinline__ void wait_async8() {
    asm volatile("s_wait_asynccnt 8" ::: "memory");   // oldest stage (8 issues) done
}

// ---------------------------------------------------------------------------
// One K=64 stage of WMMA on a 128x128 block tile.
// 8 waves as 4(M) x 2(N); wave tile 32(M) x 64(N) = 2x4 16x16 accumulators.
// As: [128 m][64 k] bf16, Bs: [128 n][64 k] bf16 (K-contiguous).
// ---------------------------------------------------------------------------
__device__ __forceinline__ void compute_stage(
    const unsigned short* AsBuf, const unsigned short* BsBuf,
    v8f (&acc)[2][4], int wm, int wn, int lm, int lh)
{
#pragma unroll
    for (int ks = 0; ks < 2; ++ks) {
        Frag16 a[2], bf[4];
#pragma unroll
        for (int mi = 0; mi < 2; ++mi) {
            int r  = wm * 32 + mi * 16 + lm;
            int ka = ks * 32 + lh * 8;               // elems 0..7 : K = lh*8+j
            a[mi].q[0] = *(const uint4*)&AsBuf[r * 64 + ka];
            a[mi].q[1] = *(const uint4*)&AsBuf[r * 64 + ka + 16]; // elems 8..15
        }
#pragma unroll
        for (int ni = 0; ni < 4; ++ni) {
            int c  = wn * 64 + ni * 16 + lm;
            int kb = ks * 32 + lh * 8;               // elems 0..7 : K = lh*8+j
            bf[ni].q[0] = *(const uint4*)&BsBuf[c * 64 + kb];
            bf[ni].q[1] = *(const uint4*)&BsBuf[c * 64 + kb + 16];
        }
#pragma unroll
        for (int mi = 0; mi < 2; ++mi)
#pragma unroll
            for (int ni = 0; ni < 4; ++ni)
                acc[mi][ni] = __builtin_amdgcn_wmma_f32_16x16x32_bf16(
                    false, a[mi].v, false, bf[ni].v, (short)0, acc[mi][ni],
                    false, false);
    }
}

#define TILE_BYTES 16384u   // 128*64*2

// ---------------------------------------------------------------------------
// GEMM 1 (per expert): hb[e] = bf16( relu(x @ W1[e] + b1[e]) * w[:,e] )
// Block 128x128, K-step 64, triple-buffered async LDS staging (depth-2).
// ---------------------------------------------------------------------------
__global__ __launch_bounds__(256) void expert_h_kernel(
    const unsigned short* __restrict__ xb,    // [B][D_IN]
    const unsigned short* __restrict__ W1t,   // [E][D_HID][D_IN] (pre-transposed)
    const float* __restrict__ b1,
    const float* __restrict__ wts,
    unsigned short* __restrict__ hb)          // [E][B][D_HID]
{
    __shared__ __align__(16) unsigned short As[3][128][64];
    __shared__ __align__(16) unsigned short Bs[3][128][64];

    const int e  = blockIdx.z;
    const int m0 = blockIdx.y * 128;
    const int n0 = blockIdx.x * 128;
    const int t  = threadIdx.x;
    const int wave = t >> 5, lane = t & 31;
    const int wm = wave >> 1, wn = wave & 1;
    const int lm = lane & 15, lh = lane >> 4;

    v8f acc[2][4];
#pragma unroll
    for (int mi = 0; mi < 2; ++mi)
#pragma unroll
        for (int ni = 0; ni < 4; ++ni)
#pragma unroll
            for (int j = 0; j < 8; ++j) acc[mi][ni][j] = 0.f;

    // Per-thread invariant DMA offsets (byte offsets)
    const unsigned asBase = (unsigned)(unsigned long long)&As[0][0][0];
    const unsigned bsBase = (unsigned)(unsigned long long)&Bs[0][0][0];
    unsigned pvA[4], pvB[4], ldsA[4], ldsB[4];
#pragma unroll
    for (int i = 0; i < 4; ++i) {
        int chunk = t + 256 * i;
        unsigned row = (unsigned)(chunk >> 3);   // 0..127
        unsigned kg  = (unsigned)(chunk & 7);    // 8 bf16 per 16B chunk
        pvA[i]  = (row * D_IN + kg * 8u) * 2u;
        pvB[i]  = (row * D_IN + kg * 8u) * 2u;   // B stride also D_IN here
        ldsA[i] = asBase + (row * 64u + kg * 8u) * 2u;
        ldsB[i] = bsBase + (row * 64u + kg * 8u) * 2u;
    }

    const unsigned short* aSt = xb  + (size_t)m0 * D_IN;                  // +s*64
    const unsigned short* bSt = W1t + ((size_t)e * D_HID + n0) * D_IN;    // +s*64

    // Prologue: stages 0 and 1 in flight
    issue4(aSt,      pvA, ldsA, 0);
    issue4(bSt,      pvB, ldsB, 0);
    issue4(aSt + 64, pvA, ldsA, TILE_BYTES);
    issue4(bSt + 64, pvB, ldsB, TILE_BYTES);
    wait_async8();          // stage 0 landed (stage 1 still in flight)
    __syncthreads();

    const int KT = D_IN / 64;   // 16 stages
    int buf = 0;
    for (int kt = 0; kt < KT; ++kt) {
        bool prefetch = (kt + 2) < KT;
        if (prefetch) {
            unsigned bo = (unsigned)((buf + 2) % 3) * TILE_BYTES;
            issue4(aSt + (kt + 2) * 64, pvA, ldsA, bo);
            issue4(bSt + (kt + 2) * 64, pvB, ldsB, bo);
        }
        compute_stage(&As[buf][0][0], &Bs[buf][0][0], acc, wm, wn, lm, lh);
        if (prefetch) wait_async8(); else wait_async0();
        __syncthreads();
        buf = (buf + 1 == 3) ? 0 : buf + 1;
    }

    // Epilogue: +b1, relu, scale by gate weight, store bf16
#pragma unroll
    for (int mi = 0; mi < 2; ++mi) {
#pragma unroll
        for (int ni = 0; ni < 4; ++ni) {
            int nn = n0 + wn * 64 + ni * 16 + lm;       // C layout: lane -> N
            float bias = b1[e * D_HID + nn];
#pragma unroll
            for (int r = 0; r < 8; ++r) {
                int mm = m0 + wm * 32 + mi * 16 + lh * 8 + r;  // vgpr r -> M
                float h = acc[mi][ni][r] + bias;
                h = h > 0.f ? h : 0.f;
                h *= wts[(size_t)mm * E_NUM + e];
                hb[((size_t)e * B_DIM + mm) * D_HID + nn] = f32_to_bf16(h);
            }
        }
    }
}

// ---------------------------------------------------------------------------
// GEMM 2: out = sum_e hb[e] @ W2[e]  (flattened K = 16*512) + sum_e w*b2
// ---------------------------------------------------------------------------
__global__ __launch_bounds__(256) void expert_out_kernel(
    const unsigned short* __restrict__ hb,    // [E][B][D_HID]
    const unsigned short* __restrict__ W2t,   // [E][D_OUT][D_HID] (pre-transposed)
    const float* __restrict__ b2,
    const float* __restrict__ wts,
    float* __restrict__ out)
{
    __shared__ __align__(16) unsigned short As[3][128][64];
    __shared__ __align__(16) unsigned short Bs[3][128][64];

    const int m0 = blockIdx.y * 128;
    const int n0 = blockIdx.x * 128;
    const int t  = threadIdx.x;
    const int wave = t >> 5, lane = t & 31;
    const int wm = wave >> 1, wn = wave & 1;
    const int lm = lane & 15, lh = lane >> 4;

    v8f acc[2][4];
#pragma unroll
    for (int mi = 0; mi < 2; ++mi)
#pragma unroll
        for (int ni = 0; ni < 4; ++ni)
#pragma unroll
            for (int j = 0; j < 8; ++j) acc[mi][ni][j] = 0.f;

    const unsigned asBase = (unsigned)(unsigned long long)&As[0][0][0];
    const unsigned bsBase = (unsigned)(unsigned long long)&Bs[0][0][0];
    unsigned pvA[4], pvB[4], ldsA[4], ldsB[4];
#pragma unroll
    for (int i = 0; i < 4; ++i) {
        int chunk = t + 256 * i;
        unsigned row = (unsigned)(chunk >> 3);
        unsigned kg  = (unsigned)(chunk & 7);
        pvA[i]  = (row * D_HID + kg * 8u) * 2u;
        pvB[i]  = (row * D_HID + kg * 8u) * 2u;
        ldsA[i] = asBase + (row * 64u + kg * 8u) * 2u;
        ldsB[i] = bsBase + (row * 64u + kg * 8u) * 2u;
    }

    // stage s (0..127): e = s>>3, k0 = (s&7)*64
    const int KT = E_NUM * (D_HID / 64);

    // Prologue: stages 0 and 1 (both e=0)
    {
        const unsigned short* aS = hb  + (size_t)m0 * D_HID;
        const unsigned short* bS = W2t + (size_t)n0 * D_HID;
        issue4(aS,      pvA, ldsA, 0);
        issue4(bS,      pvB, ldsB, 0);
        issue4(aS + 64, pvA, ldsA, TILE_BYTES);
        issue4(bS + 64, pvB, ldsB, TILE_BYTES);
        wait_async8();
        __syncthreads();
    }

    int buf = 0;
    for (int kt = 0; kt < KT; ++kt) {
        bool prefetch = (kt + 2) < KT;
        if (prefetch) {
            int s  = kt + 2;
            int es = s >> 3;
            int k0 = (s & 7) * 64;
            const unsigned short* aS = hb  + ((size_t)es * B_DIM + m0) * D_HID + k0;
            const unsigned short* bS = W2t + ((size_t)es * D_OUT + n0) * D_HID + k0;
            unsigned bo = (unsigned)((buf + 2) % 3) * TILE_BYTES;
            issue4(aS, pvA, ldsA, bo);
            issue4(bS, pvB, ldsB, bo);
        }
        compute_stage(&As[buf][0][0], &Bs[buf][0][0], acc, wm, wn, lm, lh);
        if (prefetch) wait_async8(); else wait_async0();
        __syncthreads();
        buf = (buf + 1 == 3) ? 0 : buf + 1;
    }

    // Epilogue: add rank-16 bias term sum_e w[m,e]*b2[e,n], store f32
#pragma unroll
    for (int mi = 0; mi < 2; ++mi) {
#pragma unroll
        for (int ni = 0; ni < 4; ++ni) {
            int nn = n0 + wn * 64 + ni * 16 + lm;
#pragma unroll
            for (int r = 0; r < 8; ++r) {
                int mm = m0 + wm * 32 + mi * 16 + lh * 8 + r;
                float s = acc[mi][ni][r];
                const float* wrow = wts + (size_t)mm * E_NUM;
#pragma unroll
                for (int e = 0; e < E_NUM; ++e)
                    s = fmaf(wrow[e], b2[(size_t)e * D_OUT + nn], s);
                out[(size_t)mm * D_OUT + nn] = s;
            }
        }
    }
}

// ---------------------------------------------------------------------------
extern "C" void kernel_launch(void* const* d_in, const int* in_sizes, int n_in,
                              void* d_out, int out_size, void* d_ws, size_t ws_size,
                              hipStream_t stream) {
    (void)in_sizes; (void)n_in; (void)out_size; (void)ws_size;
    const float* x  = (const float*)d_in[0];
    const float* u  = (const float*)d_in[1];
    const float* Wg = (const float*)d_in[2];
    const float* bg = (const float*)d_in[3];
    const float* W1 = (const float*)d_in[4];
    const float* b1 = (const float*)d_in[5];
    const float* W2 = (const float*)d_in[6];
    const float* b2 = (const float*)d_in[7];

    float* out = (float*)d_out;
    float* wts = out + (size_t)B_DIM * D_OUT;   // second tuple output: weights [B,E]

    // workspace: bf16 x, transposed bf16 weights, scaled hidden activations
    unsigned short* xb  = (unsigned short*)d_ws;
    unsigned short* W1t = xb  + (size_t)B_DIM * D_IN;             // [E][D_HID][D_IN]
    unsigned short* W2t = W1t + (size_t)E_NUM * D_IN  * D_HID;    // [E][D_OUT][D_HID]
    unsigned short* hb  = W2t + (size_t)E_NUM * D_HID * D_OUT;    // [E][B][D_HID]

    gate_kernel<<<B_DIM / 256, 256, 0, stream>>>(x, u, Wg, bg, wts);

    int nx = B_DIM * D_IN / 4;
    convert_kernel<<<(nx + 255) / 256, 256, 0, stream>>>(x, xb, nx);

    dim3 tb(32, 8);
    dim3 g1(D_HID / 32, D_IN / 32, E_NUM);   // W1 [E][D_IN][D_HID] -> [E][D_HID][D_IN]
    transpose_convert_kernel<<<g1, tb, 0, stream>>>(W1, W1t, D_IN, D_HID);
    dim3 g2(D_OUT / 32, D_HID / 32, E_NUM);  // W2 [E][D_HID][D_OUT] -> [E][D_OUT][D_HID]
    transpose_convert_kernel<<<g2, tb, 0, stream>>>(W2, W2t, D_HID, D_OUT);

    dim3 gA(D_HID / 128, B_DIM / 128, E_NUM);
    expert_h_kernel<<<gA, 256, 0, stream>>>(xb, W1t, b1, wts, hb);

    dim3 gB(D_OUT / 128, B_DIM / 128);
    expert_out_kernel<<<gB, 256, 0, stream>>>(hb, W2t, b2, wts, out);
}
